// UICLayer_71545565217475
// MI455X (gfx1250) — matrix operand order repaired and checked by reading, not compiled, not run on hardware
//
#include <hip/hip_runtime.h>

#define BB   128
#define TT   32
#define DD   128
#define NN   128
#define BITS 32
#define HH   4
#define CC   32
#define UU   256
#define KTOP 5
#define HD   128      // H*BITS
#define DIN  224      // 2C + BITS + D
#define O3C  96       // 3*C
#define OF   (DD + 2*HD)  // 384
#define EPSV 1e-12f

typedef __attribute__((ext_vector_type(16))) _Float16 v16h;
typedef __attribute__((ext_vector_type(2)))  _Float16 v2h;
typedef __attribute__((ext_vector_type(8)))  float    v8f;

// D-matrix store: 16x16 f32, lanes 0-15 rows 0-7, lanes 16-31 rows 8-15 (ISA 7.12.2)
__device__ __forceinline__ void stD(float* p, int ld, v8f d) {
  int lane = threadIdx.x & 31;
  int n    = lane & 15;
  int mhi  = (lane >> 4) << 3;
#pragma unroll
  for (int r = 0; r < 8; ++r) p[(r + mhi) * ld + n] = d[r];
}

__device__ __forceinline__ float sigmoidf_(float x) { return 1.0f / (1.0f + __expf(-x)); }

// per-head softmax over N=128 by one wave32; gate with g against previous w (in place)
__device__ __forceinline__ void addr_softmax(const float* logits, const float* gv, float* w, int lane) {
  float l0 = logits[lane], l1 = logits[lane + 32], l2 = logits[lane + 64], l3 = logits[lane + 96];
  float m = fmaxf(fmaxf(l0, l1), fmaxf(l2, l3));
#pragma unroll
  for (int off = 16; off; off >>= 1) m = fmaxf(m, __shfl_xor(m, off, 32));
  float e0 = __expf(l0 - m), e1 = __expf(l1 - m), e2 = __expf(l2 - m), e3 = __expf(l3 - m);
  float s = e0 + e1 + e2 + e3;
#pragma unroll
  for (int off = 16; off; off >>= 1) s += __shfl_xor(s, off, 32);
  float inv = 1.0f / s;
  int n0 = lane, n1 = lane + 32, n2 = lane + 64, n3 = lane + 96;
  w[n0] = gv[n0] * (e0 * inv) + (1.0f - gv[n0]) * w[n0];
  w[n1] = gv[n1] * (e1 * inv) + (1.0f - gv[n1]) * w[n1];
  w[n2] = gv[n2] * (e2 * inv) + (1.0f - gv[n2]) * w[n2];
  w[n3] = gv[n3] * (e3 * inv) + (1.0f - gv[n3]) * w[n3];
}

__device__ __forceinline__ float ci_val(const float* Slds, const float* Mlds, const float* xt,
                                        const float* msk, int n, int d) {
  if (d < CC) return Slds[n * CC + d];
  float mk = msk[n];
  int dd = d - CC;
  if (dd < BITS)           return mk * Mlds[n * BITS + dd];
  if (dd < BITS + CC)      return mk * Slds[n * CC + (dd - BITS)];
  return mk * xt[dd - BITS - CC];
}

constexpr int SMEM_FLOATS =
    NN * BITS + NN * CC + 3 * (HH * NN) + HD + HD + DD + NN + 8 + HH * NN +
    HD + NN + NN + UU + OF + NN * O3C + CC * O3C +
    3 * (HH * NN) + 2 * (HH * BITS) + 2 * O3C + (NN * DIN) / 2 + (DIN * O3C) / 2;

__global__ void __launch_bounds__(256, 1)
ntm_kernel(const float* __restrict__ xg,   const float* __restrict__ beta_g,
           const float* __restrict__ g_g,  const float* __restrict__ erase_g,
           const float* __restrict__ add_g,const float* __restrict__ Wg_g,
           const float* __restrict__ M0_g, const float* __restrict__ read0_g,
           const float* __restrict__ readW0_g, const float* __restrict__ writeW0_g,
           const float* __restrict__ S0_g, const float* __restrict__ Wc_g,
           const float* __restrict__ bc_g, const float* __restrict__ Wf_g,
           const float* __restrict__ bf_g, const float* __restrict__ Wi_g,
           const float* __restrict__ Ui_g, const float* __restrict__ bgru_g,
           float* __restrict__ outg)
{
  extern __shared__ float smem[];
  const int tid  = threadIdx.x;
  const int lane = tid & 31;
  const int wv   = tid >> 5;
  const int b    = blockIdx.x;

  float* p       = smem;
  float* Mlds    = p; p += NN * BITS;
  float* Slds    = p; p += NN * CC;
  float* readW   = p; p += HH * NN;
  float* writeW  = p; p += HH * NN;
  float* hisW    = p; p += HH * NN;
  float* readvec = p; p += HD;
  float* writevec= p; p += HD;
  float* xt      = p; p += DD;
  float* minv    = p; p += NN;
  float* kinv    = p; p += 8;
  float* simbuf  = p; p += HH * NN;
  float* rin     = p; p += HD;
  float* wsum    = p; p += NN;
  float* msk     = p; p += NN;
  float* hctl    = p; p += UU;
  float* obuf    = p; p += OF;
  float* xp      = p; p += NN * O3C;
  float* Uil     = p; p += CC * O3C;
  float* betaL   = p; p += HH * NN;
  float* gL      = p; p += HH * NN;
  float* WgL     = p; p += HH * NN;
  float* erL     = p; p += HH * BITS;
  float* adL     = p; p += HH * BITS;
  float* bgru    = p; p += 2 * O3C;
  _Float16* ciSW = (_Float16*)p; p += (NN * DIN) / 2;    // A-operand, fragment-swizzled
  _Float16* WiSW = (_Float16*)p; p += (DIN * O3C) / 2;   // B-operand, fragment-swizzled

  // ---- init state & constants into LDS ----
  for (int e = tid; e < NN * BITS; e += 256) Mlds[e] = M0_g[b * NN * BITS + e];
  for (int e = tid; e < NN * CC;   e += 256) Slds[e] = S0_g[b * NN * CC + e];
  for (int e = tid; e < HH * NN;   e += 256) {
    readW[e]  = readW0_g[b * HH * NN + e];
    writeW[e] = writeW0_g[b * HH * NN + e];
    hisW[e]   = 0.0f;
    betaL[e]  = beta_g[e];
    gL[e]     = g_g[e];
    WgL[e]    = Wg_g[b * HH * NN + e];
  }
  if (tid < HD) readvec[tid] = read0_g[b * HD + tid];
  if (tid < HH * BITS) { erL[tid] = erase_g[b * HH * BITS + tid]; adL[tid] = add_g[b * HH * BITS + tid]; }
  for (int e = tid; e < CC * O3C; e += 256) Uil[e] = Ui_g[e];
  if (tid < 2 * O3C) bgru[tid] = bgru_g[tid];
  // B-operand: swizzle Wi (224x96) into per-fragment lane-contiguous layout:
  // WiSW[((nt*7+kc)*32 + lane)*16 + i] = Wi[(kc*32 + ((lane>>4)<<3) + (i&7) + ((i>>3)<<4))*96 + nt*16 + (lane&15)]
  for (int e = tid; e < DIN * O3C; e += 256) {
    int i    = e & 15;
    int ln   = (e >> 4) & 31;
    int frag = e >> 9;               // 0..41
    int nt   = frag / 7, kc = frag - nt * 7;
    int k    = kc * 32 + ((ln >> 4) << 3) + (i & 7) + ((i >> 3) << 4);
    WiSW[e]  = (_Float16)Wi_g[k * O3C + nt * 16 + (ln & 15)];
  }
  __syncthreads();

  for (int t = 0; t < TT; ++t) {
    // ---- x_t + prefetch next step ----
    if (tid < DD) xt[tid] = xg[((size_t)b * TT + t) * DD + tid];
    if (t + 1 < TT) __builtin_prefetch(&xg[((size_t)b * TT + t + 1) * DD + (tid & (DD - 1))], 0, 1);

    // ---- cosine-norm factors ----
    if (tid < NN) {
      float s = 0.f;
#pragma unroll
      for (int d = 0; d < BITS; ++d) { float v = Mlds[tid * BITS + d]; s += v * v; }
      minv[tid] = rsqrtf(fmaxf(s, EPSV));
    }
    if (tid < HH) {
      float s = 0.f;
      for (int d = 0; d < BITS; ++d) { float v = readvec[tid * BITS + d]; s += v * v; }
      kinv[tid] = rsqrtf(fmaxf(s, EPSV));
    }
    __syncthreads();

    // ---- read-address logits ----
    for (int e = tid; e < HH * NN; e += 256) {
      int h = e >> 7, n = e & 127;
      float dot = 0.f;
#pragma unroll
      for (int d = 0; d < BITS; ++d) dot += readvec[h * BITS + d] * Mlds[n * BITS + d];
      simbuf[e] = betaL[e] * (-(dot * kinv[h] * minv[n]));
    }
    __syncthreads();
    if (wv < HH) addr_softmax(simbuf + wv * NN, gL + wv * NN, readW + wv * NN, lane);
    __syncthreads();

    // ---- read_input = readW @ M ; head-sum for top-k ----
    if (tid < HD) {
      int h = tid >> 5, d = tid & 31;
      float s = 0.f;
      for (int n = 0; n < NN; ++n) s += readW[h * NN + n] * Mlds[n * BITS + d];
      rin[tid] = s;
    }
    if (tid < NN) {
      wsum[tid] = readW[tid] + readW[NN + tid] + readW[2 * NN + tid] + readW[3 * NN + tid];
      msk[tid]  = 0.0f;
    }
    __syncthreads();

    // ---- top-5 mask (wave 0, register-resident, lowest-index ties) ----
    if (wv == 0) {
      float v0[4]; int id0[4];
#pragma unroll
      for (int q = 0; q < 4; ++q) { id0[q] = lane + 32 * q; v0[q] = wsum[id0[q]]; }
      for (int kk = 0; kk < KTOP; ++kk) {
        float bv = v0[0]; int bi = id0[0];
#pragma unroll
        for (int q = 1; q < 4; ++q) if (v0[q] > bv || (v0[q] == bv && id0[q] < bi)) { bv = v0[q]; bi = id0[q]; }
#pragma unroll
        for (int off = 16; off; off >>= 1) {
          float ov = __shfl_down(bv, off, 32);
          int   oi = __shfl_down(bi, off, 32);
          if (ov > bv || (ov == bv && oi < bi)) { bv = ov; bi = oi; }
        }
        bi = __shfl(bi, 0, 32);
        if (lane == 0) msk[bi] = 1.0f;
#pragma unroll
        for (int q = 0; q < 4; ++q) if (id0[q] == bi) v0[q] = -3.0e38f;
      }
    }
    __syncthreads();

    // ---- stage ci directly into A-fragment swizzled layout, packed 32-bit stores ----
    // element (n,d): mt=n>>4, kc=d>>5; lane=(n&15)+16*((d>>3)&1); idx=(d&7)+((d>>4 & 1)<<3)
    for (int e2 = tid; e2 < (NN * DIN) / 2; e2 += 256) {
      int e = e2 << 1;
      int n = e / DIN, d = e - n * DIN;            // d even; pair never straddles regions
      float f0 = ci_val(Slds, Mlds, xt, msk, n, d);
      float f1 = ci_val(Slds, Mlds, xt, msk, n, d + 1);
      int mt = n >> 4, kc = d >> 5, kk = d & 31;
      int ln = (n & 15) + (((kk >> 3) & 1) << 4);
      int idx = (kk & 7) + (((kk >> 4) & 1) << 3);
      v2h pr = { (_Float16)f0, (_Float16)f1 };
      *(v2h*)(ciSW + ((((mt * 7 + kc) * 32 + ln) << 4) + idx)) = pr;
    }
    __syncthreads();

    // ---- WMMA: xp = ci(128x224) @ Wi(224x96) + b_gru[0] ----
    for (int tile = wv; tile < 48; tile += 8) {      // wave-uniform: EXEC all ones
      int mt = tile & 7;
      int nt = tile >> 3;
      v8f acc;
      float bias = bgru[nt * 16 + (lane & 15)];
#pragma unroll
      for (int r = 0; r < 8; ++r) acc[r] = bias;
      const v16h* As = ((const v16h*)ciSW) + (mt * 7) * 32 + lane;
      const v16h* Bs = ((const v16h*)WiSW) + (nt * 7) * 32 + lane;
#pragma unroll
      for (int kc = 0; kc < 7; ++kc) {
        v16h a  = As[kc * 32];
        v16h bb = Bs[kc * 32];
        acc = __builtin_amdgcn_wmma_f32_16x16x32_f16(false, a, false, bb, (short)0, acc, false, false);
      }
      stD(xp + (mt * 16) * O3C + nt * 16, O3C, acc);
    }
    __syncthreads();

    // ---- wave 0: register/shuffle GRU over 128 slots (no barriers) ----
    // ---- waves 1-7 concurrently: hctl = relu([x_t, read] @ Wc + bc)    ----
    if (wv == 0) {
      float Uz[CC], Ur[CC], Uh[CC];
#pragma unroll
      for (int c = 0; c < CC; ++c) {
        Uz[c] = Uil[c * O3C + lane];
        Ur[c] = Uil[c * O3C + CC + lane];
        Uh[c] = Uil[c * O3C + 2 * CC + lane];
      }
      float bz = bgru[O3C + lane], br = bgru[O3C + CC + lane], bh = bgru[O3C + 2 * CC + lane];
      float hreg = 0.0f;
      for (int n = 0; n < NN; ++n) {
        float sz = bz, sr = br, sh = bh;
#pragma unroll
        for (int c = 0; c < CC; ++c) {
          float hc = __shfl(hreg, c, 32);
          sz += hc * Uz[c]; sr += hc * Ur[c]; sh += hc * Uh[c];
        }
        float z  = sigmoidf_(xp[n * O3C + lane] + sz);
        float r  = sigmoidf_(xp[n * O3C + CC + lane] + sr);
        float hcand = tanhf(xp[n * O3C + 2 * CC + lane] + r * sh);
        hreg = z * hreg + (1.0f - z) * hcand;
        Slds[n * CC + lane] = hreg;
      }
    } else {
      for (int u = tid - 32; u < UU; u += 224) {
        float s = bc_g[u];
        for (int k = 0; k < DD; ++k) s += xt[k]  * Wc_g[k * UU + u];
        for (int k = 0; k < HD; ++k) s += rin[k] * Wc_g[(DD + k) * UU + u];
        hctl[u] = fmaxf(s, 0.0f);
      }
    }
    __syncthreads();

    // ---- o = hctl @ Wf + bf (all 256 threads) ----
    for (int j = tid; j < OF; j += 256) {
      float s = bf_g[j];
      for (int u = 0; u < UU; ++u) s += hctl[u] * Wf_g[u * OF + j];
      obuf[j] = s;
    }
    __syncthreads();
    if (tid < DD)  outg[((size_t)b * TT + t) * DD + tid] = obuf[tid];
    if (tid < HD)  readvec[tid] = obuf[DD + tid];          // next pre_read
    if (tid >= HD) writevec[tid - HD] = obuf[tid + HD];    // write_head = o[256:384]
    if (tid < HH) {
      float s = 0.f;
      for (int d = 0; d < BITS; ++d) { float v = obuf[DD + HD + tid * BITS + d]; s += v * v; }
      kinv[4 + tid] = rsqrtf(fmaxf(s, EPSV));
    }
    __syncthreads();

    // ---- write addressing (pre-update M; minv still valid) ----
    for (int e = tid; e < HH * NN; e += 256) {
      int h = e >> 7, n = e & 127;
      float dot = 0.f;
#pragma unroll
      for (int d = 0; d < BITS; ++d) dot += writevec[h * BITS + d] * Mlds[n * BITS + d];
      simbuf[e] = betaL[e] * (-(dot * kinv[4 + h] * minv[n]));
    }
    __syncthreads();
    if (wv < HH) addr_softmax(simbuf + wv * NN, gL + wv * NN, writeW + wv * NN, lane);
    __syncthreads();

    // ---- M update with UNSCALED writeW ----
    for (int e = tid; e < NN * BITS; e += 256) {
      int n = e >> 5, d = e & 31;
      float ew = 0.f, aw = 0.f;
#pragma unroll
      for (int h = 0; h < HH; ++h) {
        float w = writeW[h * NN + n];
        ew += w * erL[h * BITS + d];
        aw += w * adL[h * BITS + d];
      }
      Mlds[e] = Mlds[e] * (1.0f - ew) + aw;
    }
    __syncthreads();

    // ---- Pt = softmax(Wg * hisW); writeW *= Pt; hisW += writeW ----
    if (wv < HH) {
      int h = wv;
      float l0 = WgL[h * NN + lane]      * hisW[h * NN + lane];
      float l1 = WgL[h * NN + lane + 32] * hisW[h * NN + lane + 32];
      float l2 = WgL[h * NN + lane + 64] * hisW[h * NN + lane + 64];
      float l3 = WgL[h * NN + lane + 96] * hisW[h * NN + lane + 96];
      float m = fmaxf(fmaxf(l0, l1), fmaxf(l2, l3));
#pragma unroll
      for (int off = 16; off; off >>= 1) m = fmaxf(m, __shfl_xor(m, off, 32));
      float e0 = __expf(l0 - m), e1 = __expf(l1 - m), e2 = __expf(l2 - m), e3 = __expf(l3 - m);
      float s = e0 + e1 + e2 + e3;
#pragma unroll
      for (int off = 16; off; off >>= 1) s += __shfl_xor(s, off, 32);
      float inv = 1.0f / s;
      float es[4] = { e0, e1, e2, e3 };
#pragma unroll
      for (int q = 0; q < 4; ++q) {
        int n = lane + 32 * q;
        float w = writeW[h * NN + n] * (es[q] * inv);
        writeW[h * NN + n] = w;
        hisW[h * NN + n]  += w;
      }
    }
    __syncthreads();
  }
}

extern "C" void kernel_launch(void* const* d_in, const int* in_sizes, int n_in,
                              void* d_out, int out_size, void* d_ws, size_t ws_size,
                              hipStream_t stream) {
  const float* x      = (const float*)d_in[0];
  const float* beta   = (const float*)d_in[1];
  const float* g      = (const float*)d_in[2];
  const float* erase  = (const float*)d_in[3];
  const float* add    = (const float*)d_in[4];
  const float* Wg     = (const float*)d_in[5];
  const float* M0     = (const float*)d_in[6];
  const float* read0  = (const float*)d_in[7];
  const float* readW0 = (const float*)d_in[8];
  const float* writeW0= (const float*)d_in[9];
  const float* S0     = (const float*)d_in[10];
  const float* Wc     = (const float*)d_in[11];
  const float* bc     = (const float*)d_in[12];
  const float* Wf     = (const float*)d_in[13];
  const float* bf     = (const float*)d_in[14];
  const float* Wi     = (const float*)d_in[15];
  const float* Ui     = (const float*)d_in[16];
  const float* bgru   = (const float*)d_in[17];
  float* out = (float*)d_out;

  const int smem_bytes = SMEM_FLOATS * (int)sizeof(float);
  (void)hipFuncSetAttribute((const void*)ntm_kernel,
                            hipFuncAttributeMaxDynamicSharedMemorySize, smem_bytes);
  ntm_kernel<<<dim3(BB), dim3(256), smem_bytes, stream>>>(
      x, beta, g, erase, add, Wg, M0, read0, readW0, writeW0, S0,
      Wc, bc, Wf, bf, Wi, Ui, bgru, out);
}
